// VideoClassifier_13022340841777
// MI455X (gfx1250) — matrix operand order repaired and checked by reading, not compiled, not run on hardware
//
#include <hip/hip_runtime.h>
#include <hip/hip_bf16.h>
#include <math.h>

// ---------------------------------------------------------------------------
// VideoClassifier for MI455X (gfx1250, wave32, WMMA).
// hand path: img/hand layers are linear-linear -> collapsed on device into
// Wc (1024x128) so hands_list (1.07GB) is read exactly once (HBM-bound,
// ~46us floor @ 23.3TB/s).  bf16 WMMA everywhere matrix-shaped; async
// global->LDS (ASYNCcnt) staging with double-buffered LDS pipelines the
// streams behind the WMMAs.
// ---------------------------------------------------------------------------

typedef __bf16 bf16_t;
typedef __attribute__((ext_vector_type(16))) __bf16 v16bf;
typedef __attribute__((ext_vector_type(8)))  __bf16 v8bf;
typedef __attribute__((ext_vector_type(8)))  float  v8f;
typedef int v4i_gcc __attribute__((vector_size(16)));   // matches builtin param

#define NKC   51
#define KPH   128
#define HD    128
#define FH    256
#define NVID  512
#define NTRK  4096
#define TT    32

// ------------------- gfx1250 async global->LDS path ------------------------
#if defined(__AMDGCN__) && __has_builtin(__builtin_amdgcn_global_load_async_to_lds_b128)
#define HAVE_ASYNC 1
__device__ __forceinline__ void async_copy_b128(const void* g, void* l) {
    __builtin_amdgcn_global_load_async_to_lds_b128(
        (__attribute__((address_space(1))) v4i_gcc*)(g),
        (__attribute__((address_space(3))) v4i_gcc*)(l), 0, 0);
}
__device__ __forceinline__ void wait_async0() {
#if __has_builtin(__builtin_amdgcn_s_wait_asynccnt)
    __builtin_amdgcn_s_wait_asynccnt(0);
#else
    asm volatile("s_wait_asynccnt 0" ::: "memory");
#endif
}
#else
#define HAVE_ASYNC 0
__device__ __forceinline__ void wait_async0() {}
#endif

// ---------------- WMMA fragment helpers (bf16 16x16x32) --------------------
__device__ __forceinline__ v16bf load_a_frag(const bf16_t* tile, int lane, int ld) {
    int row = lane & 15;
    int i0  = (lane & 16) ? 1 : 0;
    const v8bf* p = (const v8bf*)(tile + row * ld);
    union { v16bf v; v8bf h[2]; } u;
    u.h[0] = p[i0];
    u.h[1] = p[i0 + 2];
    return u.v;
}
__device__ __forceinline__ v16bf load_b_frag(const bf16_t* tile, int lane, int ld) {
    int col = lane & 15;
    int k0  = (lane & 16) ? 8 : 0;
    const v8bf* p = (const v8bf*)(tile + col * ld + k0);
    union { v16bf v; v8bf h[2]; } u;
    u.h[0] = p[0];
    u.h[1] = p[2];
    return u.v;
}
#define WMMA_BF16(a, b, c) \
    __builtin_amdgcn_wmma_f32_16x16x32_bf16(false, (a), false, (b), (short)0, (c), false, false)

// ---------------- ordered-uint float encoding for atomic segment-max -------
__device__ __forceinline__ unsigned fenc(float f) {
    unsigned u = __float_as_uint(f);
    return (u & 0x80000000u) ? ~u : (u | 0x80000000u);
}
__device__ __forceinline__ float fdec(unsigned e) {
    unsigned u = (e & 0x80000000u) ? (e & 0x7FFFFFFFu) : ~e;
    return __uint_as_float(u);
}

// ============================ kernel 0: init ================================
__global__ void k_init(unsigned* __restrict__ vmax, unsigned* __restrict__ vcnt) {
    int i = blockIdx.x * blockDim.x + threadIdx.x;
    if (i < NVID) { vmax[i] = 0u; vcnt[i] = 0u; }
}

// ================= kernel 1: collapse hand-encoder weights =================
__global__ __launch_bounds__(256)
void k_combine(const float* __restrict__ img_w, const float* __restrict__ img_b,
               const float* __restrict__ hand_w, const float* __restrict__ hand_b,
               bf16_t* __restrict__ Wc, float* __restrict__ bc) {
    int idx = blockIdx.x * 256 + threadIdx.x;
    int n = idx >> 10;
    int k = idx & 1023;
    float acc = 0.f;
    for (int j = 0; j < 512; ++j)
        acc += img_w[k * 512 + j] * hand_w[j * HD + n];
    Wc[(size_t)n * 1024 + k] = (bf16_t)acc;
    if (k == 0) {
        float b = hand_b[n];
        for (int j = 0; j < 512; ++j) b += img_b[j] * hand_w[j * HD + n];
        bc[n] = b;
    }
}

// ===================== kernel 2: fused hand encoder =========================
// block = one track n.  C(64x128) = bf16(hands 64x1024) x Wc(1024x128);
// max over 64 rows (T*2) + bias -> hand_feat[n][128].
// Software pipeline: A prefetched into regs (loadcnt), B staged via async
// global->LDS (asynccnt), double-buffered LDS, ONE barrier per K-chunk.
__global__ __launch_bounds__(256)
void k_hand(const float* __restrict__ hands, const bf16_t* __restrict__ Wc,
            const float* __restrict__ bc, float* __restrict__ hand_feat) {
    __shared__ __align__(16) bf16_t Alds[2][64][40];    // 2 x (64 x 32+pad)
    __shared__ __align__(16) bf16_t Blds[2][128][40];   // 2 x (128 x 32+pad)
    const int tid  = threadIdx.x;
    const int wave = tid >> 5;
    const int lane = tid & 31;
    const size_t n = blockIdx.x;

    v8f acc[4];
    #pragma unroll
    for (int mt = 0; mt < 4; ++mt) acc[mt] = (v8f)0.0f;

    const int ar = tid >> 2, ak = (tid & 3) * 8;   // A loader: row, 8-float slot
    const int br = tid >> 2, bq = tid & 3;         // B loader: row, 16B quarter

    const float* abase = hands + (n * 64 + ar) * 1024 + ak;

    // ---- prologue: chunk 0 in flight ----
    float4 af0 = ((const float4*)abase)[0];
    float4 af1 = ((const float4*)abase)[1];
#if HAVE_ASYNC
    async_copy_b128(Wc + (size_t)br * 1024 + bq * 8,        &Blds[0][br][bq * 8]);
    async_copy_b128(Wc + (size_t)(br + 64) * 1024 + bq * 8, &Blds[0][br + 64][bq * 8]);
#else
    *(v8bf*)&Blds[0][br][bq * 8]      = *(const v8bf*)(Wc + (size_t)br * 1024 + bq * 8);
    *(v8bf*)&Blds[0][br + 64][bq * 8] = *(const v8bf*)(Wc + (size_t)(br + 64) * 1024 + bq * 8);
#endif

    for (int kb = 0; kb < 32; ++kb) {
        const int buf = kb & 1;
        // commit prefetched A regs -> LDS (bf16)
        #pragma unroll
        for (int i = 0; i < 4; ++i) Alds[buf][ar][ak + i]     = (bf16_t)(&af0.x)[i];
        #pragma unroll
        for (int i = 0; i < 4; ++i) Alds[buf][ar][ak + 4 + i] = (bf16_t)(&af1.x)[i];
        wait_async0();
        __syncthreads();                       // chunk kb fully staged

        if (kb + 1 < 32) {                     // issue chunk kb+1 into buf^1
            const float* anext = abase + (kb + 1) * 32;
            __builtin_prefetch(anext + 32, 0, 0);
            af0 = ((const float4*)anext)[0];
            af1 = ((const float4*)anext)[1];
            const size_t koff = (size_t)(kb + 1) * 32 + bq * 8;
#if HAVE_ASYNC
            async_copy_b128(Wc + (size_t)br * 1024 + koff,        &Blds[buf ^ 1][br][bq * 8]);
            async_copy_b128(Wc + (size_t)(br + 64) * 1024 + koff, &Blds[buf ^ 1][br + 64][bq * 8]);
#else
            *(v8bf*)&Blds[buf ^ 1][br][bq * 8]      = *(const v8bf*)(Wc + (size_t)br * 1024 + koff);
            *(v8bf*)&Blds[buf ^ 1][br + 64][bq * 8] = *(const v8bf*)(Wc + (size_t)(br + 64) * 1024 + koff);
#endif
        }
        #pragma unroll
        for (int mt = 0; mt < 4; ++mt) {
            v16bf a = load_a_frag(&Alds[buf][mt * 16][0], lane, 40);
            v16bf b = load_b_frag(&Blds[buf][wave * 16][0], lane, 40);
            acc[mt] = WMMA_BF16(a, b, acc[mt]);
        }
    }
    float m = -INFINITY;
    #pragma unroll
    for (int mt = 0; mt < 4; ++mt)
        #pragma unroll
        for (int r = 0; r < 8; ++r) m = fmaxf(m, acc[mt][r]);
    m = fmaxf(m, __shfl_xor(m, 16, 32));       // lanes l, l^16 share a column
    const int col = wave * 16 + (lane & 15);
    if (lane < 16) hand_feat[n * HD + col] = m + bc[col];
}

// ================ kernel 3: trajectory conv1 + conv2 ========================
__global__ __launch_bounds__(128)
void k_conv(const float* __restrict__ poses,
            const float* __restrict__ w1, const float* __restrict__ b1,
            const float* __restrict__ w2, const float* __restrict__ b2,
            bf16_t* __restrict__ x2) {
    __shared__ float xin[34][NKC];
    __shared__ float y1[34][KPH];
    const int co = threadIdx.x;
    const size_t n = blockIdx.x;

    for (int e = co; e < TT * NKC; e += 128)
        xin[(e / NKC) + 1][e % NKC] = poses[n * (TT * NKC) + e];
    if (co < NKC) { xin[0][co] = 0.f; xin[33][co] = 0.f; }
    __syncthreads();

    float acc[TT];
    #pragma unroll
    for (int t = 0; t < TT; ++t) acc[t] = b1[co];
    for (int ci = 0; ci < NKC; ++ci)
        #pragma unroll
        for (int d = 0; d < 3; ++d) {
            const float w = w1[(co * NKC + ci) * 3 + d];
            #pragma unroll
            for (int t = 0; t < TT; ++t) acc[t] += w * xin[t + d][ci];
        }
    #pragma unroll
    for (int t = 0; t < TT; ++t) y1[t + 1][co] = fmaxf(acc[t], 0.f);
    y1[0][co] = 0.f; y1[33][co] = 0.f;
    __syncthreads();

    #pragma unroll
    for (int t = 0; t < TT; ++t) acc[t] = b2[co];
    for (int ci = 0; ci < KPH; ++ci)
        #pragma unroll
        for (int d = 0; d < 3; ++d) {
            const float w = w2[(co * KPH + ci) * 3 + d];
            #pragma unroll
            for (int t = 0; t < TT; ++t) acc[t] += w * y1[t + d][ci];
        }
    #pragma unroll
    for (int t = 0; t < TT; ++t)
        x2[(n * TT + t) * KPH + co] = (bf16_t)fmaxf(acc[t], 0.f);
}

// ===================== kernel 4: GRU (WMMA, 16 samples/block) ===============
// Dynamic LDS (bytes):
//   Bih [384][136] bf16 104448   @0
//   Bhh [384][136] bf16 104448   @104448
//   Xt0 [16][136]  bf16   4352   @208896
//   Xt1 [16][136]  bf16   4352   @213248
//   Ha  [16][136]  bf16   4352   @217600
//   Hf  [16][128]  f32    8192   @221952
//   gi  [16][392]  f32   25088   @230144
//   gh  [16][392]  f32   25088   @255232   -> total 280320
#define GRU_LDS_BYTES 280320
__global__ __launch_bounds__(256)
void k_gru(const bf16_t* __restrict__ x2,
           const float* __restrict__ wih, const float* __restrict__ whh,
           const float* __restrict__ bih, const float* __restrict__ bhh,
           float* __restrict__ kp_feat) {
    extern __shared__ __align__(16) char smem[];
    bf16_t* Bih = (bf16_t*)smem;
    bf16_t* Bhh = (bf16_t*)(smem + 104448);
    bf16_t* Xt0 = (bf16_t*)(smem + 208896);
    bf16_t* Xt1 = (bf16_t*)(smem + 213248);
    bf16_t* Ha  = (bf16_t*)(smem + 217600);
    float*  Hf  = (float*) (smem + 221952);
    float*  gi  = (float*) (smem + 230144);
    float*  gh  = (float*) (smem + 255232);

    const int tid  = threadIdx.x;
    const int wave = tid >> 5;
    const int lane = tid & 31;
    const size_t n0 = (size_t)blockIdx.x * 16;

    const int xs = tid >> 4, xq = tid & 15;          // x_t loader coords

    // stage GRU weights as bf16 B-operands (col-major [nr][k])
    for (int e = tid; e < 384 * KPH; e += 256) {
        int nr = e >> 7, k = e & 127;
        Bih[nr * 136 + k] = (bf16_t)wih[e];
        Bhh[nr * 136 + k] = (bf16_t)whh[e];
    }
    for (int e = tid; e < 16 * KPH; e += 256) {
        int s = e >> 7, c = e & 127;
        Ha[s * 136 + c] = (bf16_t)0.0f;
        Hf[s * 128 + c] = 0.f;
    }
    // prefetch x_0
    {
        const bf16_t* g = x2 + ((n0 + xs) * TT + 0) * KPH + xq * 8;
#if HAVE_ASYNC
        async_copy_b128(g, Xt0 + xs * 136 + xq * 8);
#else
        *(v8bf*)(Xt0 + xs * 136 + xq * 8) = *(const v8bf*)g;
#endif
    }

    const int col0 = (lane & 15);
    const int mb   = (lane & 16) ? 8 : 0;

    for (int t = 0; t < TT; ++t) {
        bf16_t* Xt = (t & 1) ? Xt1 : Xt0;
        wait_async0();
        __syncthreads();                 // x_t staged; h/gi/gh of t-1 settled

        v8f accI[3], accH[3];
        #pragma unroll
        for (int j = 0; j < 3; ++j) {
            const int nt = wave + j * 8;
            accI[j] = (v8f)0.0f; accH[j] = (v8f)0.0f;
            #pragma unroll
            for (int kc = 0; kc < 4; ++kc) {
                v16bf aX = load_a_frag(Xt + kc * 32, lane, 136);
                v16bf aH = load_a_frag(Ha + kc * 32, lane, 136);
                v16bf bI = load_b_frag(Bih + nt * 16 * 136 + kc * 32, lane, 136);
                v16bf bH = load_b_frag(Bhh + nt * 16 * 136 + kc * 32, lane, 136);
                accI[j] = WMMA_BF16(aX, bI, accI[j]);
                accH[j] = WMMA_BF16(aH, bH, accH[j]);
            }
        }
        if (t + 1 < TT) {                // prefetch x_{t+1} into other buffer
            bf16_t* Xn = (t & 1) ? Xt0 : Xt1;
            const bf16_t* g = x2 + ((n0 + xs) * TT + (t + 1)) * KPH + xq * 8;
#if HAVE_ASYNC
            async_copy_b128(g, Xn + xs * 136 + xq * 8);
#else
            *(v8bf*)(Xn + xs * 136 + xq * 8) = *(const v8bf*)g;
#endif
        }
        #pragma unroll
        for (int j = 0; j < 3; ++j) {
            const int col = (wave + j * 8) * 16 + col0;
            const float bi = bih[col], bh = bhh[col];
            #pragma unroll
            for (int r = 0; r < 8; ++r) {
                gi[(mb + r) * 392 + col] = accI[j][r] + bi;
                gh[(mb + r) * 392 + col] = accH[j][r] + bh;
            }
        }
        __syncthreads();

        for (int e = tid; e < 16 * KPH; e += 256) {
            int s = e >> 7, c = e & 127;
            float ir = gi[s * 392 + c],       hr = gh[s * 392 + c];
            float iz = gi[s * 392 + 128 + c], hz = gh[s * 392 + 128 + c];
            float in = gi[s * 392 + 256 + c], hn = gh[s * 392 + 256 + c];
            float r  = 1.f / (1.f + __expf(-(ir + hr)));
            float z  = 1.f / (1.f + __expf(-(iz + hz)));
            float nn = tanhf(in + r * hn);
            float h  = (1.f - z) * nn + z * Hf[s * 128 + c];
            Hf[s * 128 + c] = h;
            Ha[s * 136 + c] = (bf16_t)h;
        }
    }
    for (int e = tid; e < 16 * KPH; e += 256) {
        int s = e >> 7, c = e & 127;
        kp_feat[(n0 + s) * KPH + c] = Hf[s * 128 + c];
    }
}

// ============== kernel 5: track MLP + segmented max (atomic) ================
__global__ __launch_bounds__(256)
void k_mlp(const float* __restrict__ kp_feat, const float* __restrict__ hand_feat,
           const float* __restrict__ fc1w, const float* __restrict__ fc1b,
           const float* __restrict__ fc2w, const float* __restrict__ fc2b,
           const int* __restrict__ vidx,
           unsigned* __restrict__ vmax, unsigned* __restrict__ vcnt) {
    __shared__ float trk[FH];
    __shared__ float red[FH];
    const int j = threadIdx.x;
    const size_t n = blockIdx.x;
    trk[j] = (j < KPH) ? kp_feat[n * KPH + j] : hand_feat[n * HD + (j - KPH)];
    __syncthreads();
    float h = fc1b[j];
    for (int k = 0; k < FH; ++k) h += trk[k] * fc1w[k * FH + j];
    red[j] = fmaxf(h, 0.f) * fc2w[j];
    __syncthreads();
    for (int s = 128; s > 0; s >>= 1) {
        if (j < s) red[j] += red[j + s];
        __syncthreads();
    }
    if (j == 0) {
        const float logit = red[0] + fc2b[0];
        const int v = vidx[n];
        atomicMax(&vmax[v], fenc(logit));
        atomicAdd(&vcnt[v], 1u);
    }
}

// ======================= kernel 6: finalize =================================
__global__ void k_final(const unsigned* __restrict__ vmax, const unsigned* __restrict__ vcnt,
                        const float* __restrict__ nts, float* __restrict__ out) {
    int i = blockIdx.x * blockDim.x + threadIdx.x;
    if (i < NVID) {
        float logit = (vcnt[i] > 0u) ? fdec(vmax[i]) : nts[0];
        out[i] = 1.f / (1.f + __expf(-logit));
    }
}

// ============================================================================
extern "C" void kernel_launch(void* const* d_in, const int* in_sizes, int n_in,
                              void* d_out, int out_size, void* d_ws, size_t ws_size,
                              hipStream_t stream) {
    const float* poses  = (const float*)d_in[0];
    const float* hands  = (const float*)d_in[1];
    const int*   vidx   = (const int*)  d_in[2];
    const float* w1     = (const float*)d_in[3];
    const float* b1     = (const float*)d_in[4];
    const float* w2     = (const float*)d_in[5];
    const float* b2     = (const float*)d_in[6];
    const float* wih    = (const float*)d_in[7];
    const float* whh    = (const float*)d_in[8];
    const float* bihp   = (const float*)d_in[9];
    const float* bhhp   = (const float*)d_in[10];
    const float* imgw   = (const float*)d_in[11];
    const float* imgb   = (const float*)d_in[12];
    const float* handw  = (const float*)d_in[13];
    const float* handb  = (const float*)d_in[14];
    const float* fc1w   = (const float*)d_in[15];
    const float* fc1b   = (const float*)d_in[16];
    const float* fc2w   = (const float*)d_in[17];
    const float* fc2b   = (const float*)d_in[18];
    const float* nts    = (const float*)d_in[19];
    float* out = (float*)d_out;

    char* ws = (char*)d_ws;
    bf16_t*   Wc        = (bf16_t*)  (ws + 0);            // 262144
    float*    bc        = (float*)   (ws + 262144);       // 512
    float*    hand_feat = (float*)   (ws + 262656);       // 2097152
    float*    kp_feat   = (float*)   (ws + 2359808);      // 2097152
    bf16_t*   x2        = (bf16_t*)  (ws + 4456960);      // 33554432
    unsigned* vmax      = (unsigned*)(ws + 38011392);     // 2048
    unsigned* vcnt      = (unsigned*)(ws + 38013440);     // 2048

    k_init   <<<2,   256, 0, stream>>>(vmax, vcnt);
    k_combine<<<512, 256, 0, stream>>>(imgw, imgb, handw, handb, Wc, bc);
    k_hand   <<<NTRK, 256, 0, stream>>>(hands, Wc, bc, hand_feat);
    k_conv   <<<NTRK, 128, 0, stream>>>(poses, w1, b1, w2, b2, x2);
    k_gru    <<<NTRK / 16, 256, GRU_LDS_BYTES, stream>>>(x2, wih, whh, bihp, bhhp, kp_feat);
    k_mlp    <<<NTRK, 256, 0, stream>>>(kp_feat, hand_feat, fc1w, fc1b, fc2w, fc2b,
                                        vidx, vmax, vcnt);
    k_final  <<<2,   256, 0, stream>>>(vmax, vcnt, nts, out);
}
